// PointtadDecoderHead_59347858096823
// MI455X (gfx1250) — matrix operand order repaired and checked by reading, not compiled, not run on hardware
//
#include <hip/hip_runtime.h>

// ---------------- static dims ----------------
constexpr int kN = 32, kNR = 64, kT = 512, kD = 256, kP = 32, kS = 4, kNB = 4;
constexpr int kDFF = 2048, kNH = 8, kNC = 200, kDYN = 64;
constexpr int kB = kN * kNR;            // 2048
constexpr float kWIN = 512.0f;

// ---------------- WMMA types ----------------
typedef __attribute__((ext_vector_type(16))) _Float16 v16h;
typedef __attribute__((ext_vector_type(8)))  float    v8f;
typedef __attribute__((ext_vector_type(4)))  unsigned int u32x4;
typedef __attribute__((ext_vector_type(8)))  int    i32x8;
typedef __attribute__((ext_vector_type(4)))  int    i32x4;

union FragU { uint4 u[2]; v16h h; };
union FragE { _Float16 e[16]; v16h h; };

__device__ __forceinline__ v8f vzero8() {
  v8f z = {0.f,0.f,0.f,0.f,0.f,0.f,0.f,0.f};
  return z;
}

__device__ __forceinline__ v8f wmma_f16(v16h a, v16h b, v8f c) {
  return __builtin_amdgcn_wmma_f32_16x16x32_f16(
      false, a, false, b, (short)0, c, false, false);
}

// A fragment: 16x32 from row-major source (used by the batched dyn-mix kernels).
__device__ __forceinline__ v16h load_a_frag(const _Float16* base, size_t ld) {
  int lane = threadIdx.x & 31;
  int row = lane & 15;
  const _Float16* p = base + (size_t)row * ld + ((lane & 16) ? 8 : 0);
  FragU f;
  f.u[0] = *reinterpret_cast<const uint4*>(p);
  f.u[1] = *reinterpret_cast<const uint4*>(p + 16);
  return f.h;
}

// B fragment from a true K x N matrix with element (k,n) at base[k*kstride + n].
__device__ __forceinline__ v16h load_b_strided(const _Float16* base, size_t kstride) {
  int lane = threadIdx.x & 31;
  int c = lane & 15;
  int k0 = (lane & 16) ? 8 : 0;
  FragE f;
#pragma unroll
  for (int i = 0; i < 16; ++i) f.e[i] = base[(size_t)(k0 + i + ((i >= 8) ? 8 : 0)) * kstride + c];
  return f.h;
}

// ---------------- LDS fragment loads (padded row stride 40 halves = 80B, bank-conflict free) ----
constexpr int kLdsPitch = 40;  // halves per row (32 data + 8 pad)

__device__ __forceinline__ v16h lds_a_frag(const _Float16* lA, int mrow) {
  int lane = threadIdx.x & 31;
  const _Float16* p = lA + (size_t)(mrow + (lane & 15)) * kLdsPitch + ((lane & 16) ? 8 : 0);
  FragU f;
  f.u[0] = *reinterpret_cast<const uint4*>(p);
  f.u[1] = *reinterpret_cast<const uint4*>(p + 16);
  return f.h;
}

__device__ __forceinline__ v16h lds_b_frag(const _Float16* lB, int ncol) {
  int lane = threadIdx.x & 31;
  const _Float16* p = lB + (size_t)(ncol + (lane & 15)) * kLdsPitch + ((lane & 16) ? 8 : 0);
  FragU f;
  f.u[0] = *reinterpret_cast<const uint4*>(p);
  f.u[1] = *reinterpret_cast<const uint4*>(p + 16);
  return f.h;
}

// ---------------- Tensor Data Mover: stage a 32(k) x 64(n) f16 weight tile into LDS -------------
// W is (Nn, K) row-major halves; tile starts at row n0, col k. TDM pads LDS dest with
// 4 DWORDs every 16 DWORDs -> 80B row pitch (matches kLdsPitch). Rows >= Nn zero-fill.
__device__ __forceinline__ void tdm_load_w_tile(const _Float16* W, unsigned ldsOff,
                                                int n0, int Nn, int K, int k) {
  unsigned long long gaddr =
      (unsigned long long)(size_t)W + ((unsigned long long)n0 * (unsigned)K + (unsigned)k) * 2ull;
  u32x4 g0;
  g0[0] = 0x1u;                                   // count=1, user descriptor
  g0[1] = ldsOff;                                 // lds_addr (bytes)
  g0[2] = (unsigned)(gaddr & 0xffffffffu);        // global_addr[31:0]
  g0[3] = (unsigned)((gaddr >> 32) & 0x1ffffffu)  // global_addr[56:32]
        | (2u << 30);                             // type = 2 (image)
  unsigned td0 = (unsigned)K;                     // tensor_dim0 (elements)
  unsigned td1 = (unsigned)Nn;                    // tensor_dim1 (rows)
  i32x8 g1;
  g1[0] = (int)((1u << 16)        // data_size = 1 (2 bytes)
              | (1u << 20)        // pad_enable
              | (3u << 22)        // pad_interval = 16 DWORDs
              | (3u << 25));      // pad_amount   = 4 DWORDs
  g1[1] = (int)((td0 & 0xffffu) << 16);
  g1[2] = (int)(((td0 >> 16) & 0xffffu) | ((td1 & 0xffffu) << 16));
  g1[3] = (int)(((td1 >> 16) & 0xffffu) | (32u << 16));   // tile_dim0 = 32
  g1[4] = (int)64;                                        // tile_dim1 = 64, tile_dim2 = 0
  g1[5] = (int)(unsigned)(td0);                           // tensor_dim0_stride[31:0] = K
  g1[6] = 0;                                              // stride hi, dim1_stride lo
  g1[7] = 0;
  i32x4 z4 = {0, 0, 0, 0};
#if defined(__clang_major__) && (__clang_major__ >= 23)
  i32x8 z8 = {0, 0, 0, 0, 0, 0, 0, 0};
  __builtin_amdgcn_tensor_load_to_lds(g0, g1, z4, z4, z8, 0);
#else
  __builtin_amdgcn_tensor_load_to_lds(g0, g1, z4, z4, 0);
#endif
}

// ---------------- generic WMMA GEMM: out = X(M,K) @ W(N,K)^T (+bias)(+relu) --------------------
// Block 128 threads (4 waves, 2x2), 64x64 block tile, BK=32, double-buffered LDS:
//   A tile staged by cooperative global_load_b128 + ds_store_b128,
//   W tile staged by the Tensor Data Mover (wave 0 issues, s_wait_tensorcnt, barrier).
__global__ void __launch_bounds__(128) gemm_wmma_lds_k(
    const _Float16* __restrict__ X, const _Float16* __restrict__ W,
    const float* __restrict__ bias,
    float* __restrict__ outF, _Float16* __restrict__ outH,
    int M, int Nn, int K, int relu)
{
  __shared__ _Float16 lA[2][64 * kLdsPitch];
  __shared__ _Float16 lB[2][64 * kLdsPitch];

  int wave = threadIdx.x >> 5;
  int lane = threadIdx.x & 31;
  int m0 = blockIdx.y * 64;
  int n0 = blockIdx.x * 64;
  int wm = (wave >> 1) * 32;
  int wn = (wave & 1) * 32;
  int arow = threadIdx.x >> 2;       // 0..31 (x2 passes) for A staging
  int aseg = threadIdx.x & 3;

  v8f acc[2][2];
#pragma unroll
  for (int i = 0; i < 2; ++i)
#pragma unroll
    for (int j = 0; j < 2; ++j) acc[i][j] = vzero8();

  int nk = K >> 5;

  // ---- prologue: stage tile 0 ----
#pragma unroll
  for (int r = 0; r < 2; ++r) {
    int row = arow + r * 32;
    uint4 v = *reinterpret_cast<const uint4*>(X + (size_t)(m0 + row) * K + aseg * 8);
    *reinterpret_cast<uint4*>(&lA[0][row * kLdsPitch + aseg * 8]) = v;
  }
  if (wave == 0) {
    tdm_load_w_tile(W, (unsigned)(size_t)&lB[0][0], n0, Nn, K, 0);
    __builtin_amdgcn_s_wait_tensorcnt(0);
  }
  __syncthreads();

  for (int kt = 0; kt < nk; ++kt) {
    int cb = kt & 1;
    int nb = cb ^ 1;
    bool havenext = (kt + 1) < nk;
    uint4 pre[2];
    if (havenext) {
      int knxt = (kt + 1) << 5;
#pragma unroll
      for (int r = 0; r < 2; ++r) {
        int row = arow + r * 32;
        pre[r] = *reinterpret_cast<const uint4*>(X + (size_t)(m0 + row) * K + knxt + aseg * 8);
      }
      if (wave == 0)
        tdm_load_w_tile(W, (unsigned)(size_t)&lB[nb][0], n0, Nn, K, knxt);
    }
    if (kt + 2 < nk)  // L2 warm hint two steps ahead
      __builtin_prefetch(X + (size_t)(m0 + arow) * K + ((kt + 2) << 5), 0, 1);

    // ---- compute from current buffers ----
    v16h a0 = lds_a_frag(lA[cb], wm);
    v16h a1 = lds_a_frag(lA[cb], wm + 16);
    v16h b0 = lds_b_frag(lB[cb], wn);
    v16h b1 = lds_b_frag(lB[cb], wn + 16);
    acc[0][0] = wmma_f16(a0, b0, acc[0][0]);
    acc[0][1] = wmma_f16(a0, b1, acc[0][1]);
    acc[1][0] = wmma_f16(a1, b0, acc[1][0]);
    acc[1][1] = wmma_f16(a1, b1, acc[1][1]);

    if (havenext) {
#pragma unroll
      for (int r = 0; r < 2; ++r) {
        int row = arow + r * 32;
        *reinterpret_cast<uint4*>(&lA[nb][row * kLdsPitch + aseg * 8]) = pre[r];
      }
      if (wave == 0) __builtin_amdgcn_s_wait_tensorcnt(0);
    }
    __syncthreads();
  }

  // ---- epilogue ----
#pragma unroll
  for (int mi = 0; mi < 2; ++mi)
#pragma unroll
    for (int ni = 0; ni < 2; ++ni) {
      int col = n0 + wn + ni * 16 + (lane & 15);
      float bv = 0.f;
      if (bias && col < Nn) bv = bias[col];
#pragma unroll
      for (int i = 0; i < 8; ++i) {
        int r = m0 + wm + mi * 16 + i + ((lane & 16) ? 8 : 0);
        if (r < M && col < Nn) {
          float o = acc[mi][ni][i] + bv;
          if (relu) o = fmaxf(o, 0.f);
          if (outF) outF[(size_t)r * Nn + col] = o;
          if (outH) outH[(size_t)r * Nn + col] = (_Float16)o;
        }
      }
    }
}

// ---------------- f32 -> f16 convert ----------------
__global__ void cvt_k(const float* __restrict__ src, _Float16* __restrict__ dst, int n) {
  int i = blockIdx.x * 256 + threadIdx.x;
  if (i < n) dst[i] = (_Float16)src[i];
}

// ---------------- point deformation: small projections + softmax over S ----------------
__global__ void __launch_bounds__(32) pd_proj_k(
    const float* __restrict__ pro,
    const float* boff_w, const float* boff_b,
    const float* coff_w, const float* coff_b,
    const float* bw_w, const float* bw_b,
    const float* cw_w, const float* cw_b,
    float* __restrict__ pdout)
{
  int b = blockIdx.x;
  int lane = threadIdx.x;
  if (lane >= 16) return;
  int w = lane >> 2, s = lane & 3;
  const float* wm = (w == 0) ? boff_w : (w == 1) ? coff_w : (w == 2) ? bw_w : cw_w;
  const float* wb = (w == 0) ? boff_b : (w == 1) ? coff_b : (w == 2) ? bw_b : cw_b;
  const float* x = pro + (size_t)b * kD;
  const float* wr = wm + s * kD;
  float acc = wb[s];
  for (int k = 0; k < kD; ++k) acc += x[k] * wr[k];
  float outv = acc;
  if (w >= 2) {
    float m = acc;
    m = fmaxf(m, __shfl_xor(m, 1, 4));
    m = fmaxf(m, __shfl_xor(m, 2, 4));
    float e = expf(acc - m);
    float sum = e;
    sum += __shfl_xor(sum, 1, 4);
    sum += __shfl_xor(sum, 2, 4);
    outv = e / sum;
  }
  pdout[(size_t)b * 16 + w * 4 + s] = outv;
}

// ---------------- 1-D bilinear sampling + weighted sum over subpoints ----------------
__global__ void __launch_bounds__(256) pd_sample_k(
    const float* __restrict__ features, const float* __restrict__ prop,
    const float* __restrict__ pdout,
    float* __restrict__ pf, _Float16* __restrict__ pfh)
{
  int b = blockIdx.x, p = blockIdx.y, d = threadIdx.x;
  int n = b >> 6;
  float pt = prop[(size_t)b * kP + p];
  const float* pdo = pdout + (size_t)b * 16;
  float acc = 0.f;
#pragma unroll
  for (int s = 0; s < kS; ++s) {
    float off = (p < kNB) ? pdo[s] : pdo[4 + s];
    float wt  = (p < kNB) ? pdo[8 + s] : pdo[12 + s];
    float loc = fminf(fmaxf(pt + off * (1.0f / kWIN), 0.f), 1.f);
    float px = loc * (float)kT - 0.5f;
    float fl = floorf(px);
    int i0 = (int)fl;
    float w = px - fl;
    int i1 = i0 + 1;
    float v0 = (i0 >= 0 && i0 < kT) ? features[((size_t)n * kT + i0) * kD + d] : 0.f;
    float v1 = (i1 >= 0 && i1 < kT) ? features[((size_t)n * kT + i1) * kD + d] : 0.f;
    acc += wt * ((1.f - w) * v0 + w * v1);
  }
  size_t idx = ((size_t)p * kB + b) * kD + d;
  pf[idx] = acc;
  pfh[idx] = (_Float16)acc;
}

// ---------------- self-attention, per (n, head) ----------------
__global__ void __launch_bounds__(64) attn_k(const _Float16* __restrict__ qkvh,
                                             _Float16* __restrict__ oah)
{
  int nh = blockIdx.x;
  int n = nh / kNH, h = nh % kNH;
  int l = threadIdx.x;
  const float scale = 0.17677669529663687f;
  float q[32];
#pragma unroll
  for (int j = 0; j < 32; ++j)
    q[j] = (float)qkvh[(size_t)(n * kNR + l) * (3 * kD) + h * 32 + j];
  float mx = -1e30f, sum = 0.f;
  for (int m = 0; m < kNR; ++m) {
    const _Float16* kp = qkvh + (size_t)(n * kNR + m) * (3 * kD) + kD + h * 32;
    float acc = 0.f;
#pragma unroll
    for (int j = 0; j < 32; ++j) acc += q[j] * (float)kp[j];
    acc *= scale;
    float nmx = fmaxf(mx, acc);
    sum = sum * expf(mx - nmx) + expf(acc - nmx);
    mx = nmx;
  }
  float inv = 1.f / sum;
  float o[32];
#pragma unroll
  for (int j = 0; j < 32; ++j) o[j] = 0.f;
  for (int m = 0; m < kNR; ++m) {
    const _Float16* kp = qkvh + (size_t)(n * kNR + m) * (3 * kD) + kD + h * 32;
    float acc = 0.f;
#pragma unroll
    for (int j = 0; j < 32; ++j) acc += q[j] * (float)kp[j];
    float pm = expf(acc * scale - mx) * inv;
    const _Float16* vp = qkvh + (size_t)(n * kNR + m) * (3 * kD) + 2 * kD + h * 32;
#pragma unroll
    for (int j = 0; j < 32; ++j) o[j] += pm * (float)vp[j];
  }
#pragma unroll
  for (int j = 0; j < 32; ++j)
    oah[(size_t)(n * kNR + l) * kD + h * 32 + j] = (_Float16)o[j];
}

// ---------------- row LayerNorm (+optional residual, relu, f32/f16 outs) ----------------
__global__ void __launch_bounds__(256) ln_row_k(
    const float* __restrict__ x, const float* __restrict__ res,
    const float* __restrict__ w, const float* __restrict__ bvec,
    float* __restrict__ outF, _Float16* __restrict__ outH,
    int L, int ldH, int relu)
{
  __shared__ float red[256];
  int row = blockIdx.x, t = threadIdx.x;
  float xv = 0.f;
  if (t < L) {
    xv = x[(size_t)row * L + t];
    if (res) xv += res[(size_t)row * L + t];
  }
  red[t] = (t < L) ? xv : 0.f;
  __syncthreads();
  for (int s = 128; s > 0; s >>= 1) { if (t < s) red[t] += red[t + s]; __syncthreads(); }
  float mean = red[0] / (float)L;
  __syncthreads();
  float dd = (t < L) ? (xv - mean) : 0.f;
  red[t] = dd * dd;
  __syncthreads();
  for (int s = 128; s > 0; s >>= 1) { if (t < s) red[t] += red[t + s]; __syncthreads(); }
  float rs = rsqrtf(red[0] / (float)L + 1e-5f);
  if (t < L) {
    float o = (xv - mean) * rs * w[t] + bvec[t];
    if (relu) o = fmaxf(o, 0.f);
    if (outF) outF[(size_t)row * L + t] = o;
    if (outH) outH[(size_t)row * ldH + t] = (_Float16)o;
  }
}

// ---------------- diff path: G[b][q][d] = sum_p tparam[b][p][q] * pf[p][b][d] ----------------
__global__ void __launch_bounds__(256) diff_gemm_k(const float* __restrict__ tdyn,
                                                   const float* __restrict__ pf,
                                                   float* __restrict__ G)
{
  int b = blockIdx.x, d = threadIdx.x;
  float fv[32];
#pragma unroll
  for (int p = 0; p < kP; ++p) fv[p] = pf[((size_t)p * kB + b) * kD + d];
  const float* tp = tdyn + (size_t)b * (kP * kP);
  for (int q = 0; q < kP; ++q) {
    float acc = 0.f;
#pragma unroll
    for (int p = 0; p < kP; ++p) acc += tp[p * kP + q] * fv[p];
    G[(size_t)b * (kP * kD) + (size_t)q * kD + d] = acc;
  }
}

__global__ void __launch_bounds__(256) diff_ln_k(const float* __restrict__ G,
                                                 const float* __restrict__ n4w,
                                                 const float* __restrict__ n4b,
                                                 _Float16* __restrict__ fcat)
{
  int b = blockIdx.x, d = threadIdx.x;
  const float* g = G + (size_t)b * (kP * kD) + d;
  float vals[32];
  float m = 0.f;
#pragma unroll
  for (int p = 0; p < kP; ++p) { vals[p] = g[(size_t)p * kD]; m += vals[p]; }
  m *= (1.f / kP);
  float v = 0.f;
#pragma unroll
  for (int p = 0; p < kP; ++p) { float t = vals[p] - m; v += t * t; }
  float rs = rsqrtf(v * (1.f / kP) + 1e-5f);
#pragma unroll
  for (int p = 0; p < kP; ++p) {
    float o = (vals[p] - m) * rs * n4w[p] + n4b[p];
    o = fmaxf(o, 0.f);
    fcat[(size_t)b * (2 * kD * kP) + (size_t)p * (2 * kD) + kD + d] = (_Float16)o;
  }
}

// ------- batched dynamic conv 1: f1[b] (P x DYN) = feats[b] (P x D) @ param1[b] (D x DYN) -------
__global__ void __launch_bounds__(64) dyn_mix1_k(const _Float16* __restrict__ pfh,
                                                 const _Float16* __restrict__ dynh,
                                                 float* __restrict__ f1)
{
  int b = blockIdx.x;
  int wave = threadIdx.x >> 5, lane = threadIdx.x & 31;
  const size_t ldA = (size_t)kB * kD;
  const _Float16* Ab = pfh + (size_t)b * kD;
  const _Float16* Bb = dynh + (size_t)b * (2 * kD * kDYN);
  v8f acc[2][2];
#pragma unroll
  for (int i = 0; i < 2; ++i)
#pragma unroll
    for (int j = 0; j < 2; ++j) acc[i][j] = vzero8();
  for (int k = 0; k < kD; k += 32) {
    v16h a0 = load_a_frag(Ab + k, ldA);
    v16h a1 = load_a_frag(Ab + 16 * ldA + k, ldA);
    int n0 = wave * 32;
    v16h b0 = load_b_strided(Bb + (size_t)k * kDYN + n0, kDYN);
    v16h b1 = load_b_strided(Bb + (size_t)k * kDYN + n0 + 16, kDYN);
    acc[0][0] = wmma_f16(a0, b0, acc[0][0]);
    acc[0][1] = wmma_f16(a0, b1, acc[0][1]);
    acc[1][0] = wmma_f16(a1, b0, acc[1][0]);
    acc[1][1] = wmma_f16(a1, b1, acc[1][1]);
  }
  float* out = f1 + (size_t)b * (kP * kDYN);
#pragma unroll
  for (int mi = 0; mi < 2; ++mi)
#pragma unroll
    for (int ni = 0; ni < 2; ++ni) {
      int col = wave * 32 + ni * 16 + (lane & 15);
#pragma unroll
      for (int i = 0; i < 8; ++i) {
        int p = mi * 16 + i + ((lane & 16) ? 8 : 0);
        out[(size_t)p * kDYN + col] = acc[mi][ni][i];
      }
    }
}

// ------- batched dynamic conv 2: f2[b] (P x D) = f1p[b] (P x DYN) @ param2[b] (DYN x D) -------
__global__ void __launch_bounds__(256) dyn_mix2_k(const _Float16* __restrict__ f1ph,
                                                  const _Float16* __restrict__ dynh,
                                                  float* __restrict__ f2)
{
  int b = blockIdx.x;
  int wave = threadIdx.x >> 5, lane = threadIdx.x & 31;
  const _Float16* Ab = f1ph + (size_t)b * (kP * kDYN);
  const _Float16* Bb = dynh + (size_t)b * (2 * kD * kDYN) + (size_t)kD * kDYN;
  int n0w = wave * 32;
  v8f acc[2][2];
#pragma unroll
  for (int i = 0; i < 2; ++i)
#pragma unroll
    for (int j = 0; j < 2; ++j) acc[i][j] = vzero8();
  for (int k = 0; k < kDYN; k += 32) {
    v16h a0 = load_a_frag(Ab + k, kDYN);
    v16h a1 = load_a_frag(Ab + 16 * kDYN + k, kDYN);
    v16h b0 = load_b_strided(Bb + (size_t)k * kD + n0w, kD);
    v16h b1 = load_b_strided(Bb + (size_t)k * kD + n0w + 16, kD);
    acc[0][0] = wmma_f16(a0, b0, acc[0][0]);
    acc[0][1] = wmma_f16(a0, b1, acc[0][1]);
    acc[1][0] = wmma_f16(a1, b0, acc[1][0]);
    acc[1][1] = wmma_f16(a1, b1, acc[1][1]);
  }
  float* out = f2 + (size_t)b * (kP * kD);
#pragma unroll
  for (int mi = 0; mi < 2; ++mi)
#pragma unroll
    for (int ni = 0; ni < 2; ++ni) {
      int col = n0w + ni * 16 + (lane & 15);
#pragma unroll
      for (int i = 0; i < 8; ++i) {
        int p = mi * 16 + i + ((lane & 16) ? 8 : 0);
        out[(size_t)p * kD + col] = acc[mi][ni][i];
      }
    }
}

// ---------------- pred points: prop + deltas / WIN ----------------
__global__ void pred_k(const float* __restrict__ prop, const float* __restrict__ deltas,
                       float* __restrict__ out)
{
  int i = blockIdx.x * 256 + threadIdx.x;
  if (i < kB * kP) out[i] = prop[i] + deltas[i] * (1.0f / kWIN);
}

// ---------------- input indices ----------------
enum {
  IN_FEATURES = 0, IN_PROPOSAL, IN_PRO,
  IN_PD_BOFF_W, IN_PD_BOFF_B, IN_PD_COFF_W, IN_PD_COFF_B,
  IN_PD_BW_W, IN_PD_BW_B, IN_PD_CW_W, IN_PD_CW_B,
  IN_ATTN_IN_W, IN_ATTN_IN_B, IN_ATTN_OUT_W, IN_ATTN_OUT_B,
  IN_NORM1_W, IN_NORM1_B,
  IN_PM_DYN_W, IN_PM_DYN_B, IN_PM_TDYN_W, IN_PM_TDYN_B,
  IN_PM_N4_W, IN_PM_N4_B, IN_PM_N1_W, IN_PM_N1_B, IN_PM_N2_W, IN_PM_N2_B,
  IN_PM_OUT_W, IN_PM_OUT_B, IN_PM_N3_W, IN_PM_N3_B,
  IN_NORM2_W, IN_NORM2_B,
  IN_LIN1_W, IN_LIN1_B, IN_LIN2_W, IN_LIN2_B,
  IN_NORM3_W, IN_NORM3_B,
  IN_CLS_W, IN_CLS_LN_W, IN_CLS_LN_B,
  IN_REG_W, IN_REG_LN_W, IN_REG_LN_B,
  IN_LOGITS_W, IN_LOGITS_B, IN_DELTA_W, IN_DELTA_B
};

static inline void launch_gemm(const _Float16* X, const _Float16* W, const float* bias,
                               float* outF, _Float16* outH, int M, int Nn, int K,
                               int relu, hipStream_t s) {
  dim3 g((Nn + 63) / 64, (M + 63) / 64);
  gemm_wmma_lds_k<<<g, 128, 0, s>>>(X, W, bias, outF, outH, M, Nn, K, relu);
}

extern "C" void kernel_launch(void* const* d_in, const int* in_sizes, int n_in,
                              void* d_out, int out_size, void* d_ws, size_t ws_size,
                              hipStream_t stream) {
  (void)in_sizes; (void)n_in; (void)out_size; (void)ws_size;
  char* ws = (char*)d_ws;
  size_t cur = 0;
  auto alloc = [&](size_t bytes) -> size_t {
    size_t o = cur;
    cur += (bytes + 255) & ~(size_t)255;
    return o;
  };

  size_t o_h_ain  = alloc((size_t)3 * kD * kD * 2);
  size_t o_h_aout = alloc((size_t)kD * kD * 2);
  size_t o_h_dyn  = alloc((size_t)2 * kD * kDYN * kD * 2);
  size_t o_h_tdyn = alloc((size_t)kP * kP * kD * 2);
  size_t o_h_pout = alloc((size_t)kD * 2 * kD * kP * 2);
  size_t o_h_lin1 = alloc((size_t)kDFF * kD * 2);
  size_t o_h_lin2 = alloc((size_t)kD * kDFF * 2);
  size_t o_h_cls  = alloc((size_t)kD * kD * 2);
  size_t o_h_reg  = alloc((size_t)3 * kD * kD * 2);
  size_t o_h_logi = alloc((size_t)(kNC + 1) * kD * 2);
  size_t o_h_dlt  = alloc((size_t)kP * kD * 2);
  size_t o_proh  = alloc((size_t)kB * kD * 2);
  size_t o_pdout = alloc((size_t)kB * 16 * 4);
  size_t o_pf    = alloc((size_t)kP * kB * kD * 4);
  size_t o_pfh   = alloc((size_t)kP * kB * kD * 2);
  size_t o_qkvh  = alloc((size_t)kB * 3 * kD * 2);
  size_t o_oah   = alloc((size_t)kB * kD * 2);
  size_t o_oproj = alloc((size_t)kB * kD * 4);
  size_t o_x1    = alloc((size_t)kB * kD * 4);
  size_t o_x1h   = alloc((size_t)kB * kD * 2);
  size_t o_dynh  = alloc((size_t)kB * 2 * kD * kDYN * 2);
  size_t o_tdyn  = alloc((size_t)kB * kP * kP * 4);
  size_t o_G     = alloc((size_t)kB * kP * kD * 4);
  size_t o_f1    = alloc((size_t)kB * kP * kDYN * 4);
  size_t o_f1ph  = alloc((size_t)kB * kP * kDYN * 2);
  size_t o_pmout = alloc((size_t)kB * kD * 4);
  size_t o_cfeat = alloc((size_t)kB * kD * 4);
  size_t o_act1  = alloc((size_t)kB * kD * 4);
  size_t o_act1h = alloc((size_t)kB * kD * 2);
  size_t o_hff   = alloc((size_t)kB * kDFF * 2);
  size_t o_ff    = alloc((size_t)kB * kD * 4);
  size_t o_fch   = alloc((size_t)kB * kD * 2);
  size_t o_clsr  = alloc((size_t)kB * kD * 4);
  size_t o_clsfh = alloc((size_t)kB * kD * 2);
  size_t o_regF  = alloc((size_t)kB * kD * 4);
  size_t o_regh0 = alloc((size_t)kB * kD * 2);
  size_t o_regh1 = alloc((size_t)kB * kD * 2);
  size_t o_dlt   = alloc((size_t)kB * kP * 4);
  size_t o_fcat = o_pf;   // alias: pf(f32) dead after diff_gemm
  size_t o_f2   = o_G;    // alias: G dead after diff_ln

#define WF(off) reinterpret_cast<float*>(ws + (off))
#define WH(off) reinterpret_cast<_Float16*>(ws + (off))
  auto inf = [&](int i) { return (const float*)d_in[i]; };
  float* dout = (float*)d_out;

  auto cvt = [&](int idx, size_t dsto, int n) {
    cvt_k<<<(n + 255) / 256, 256, 0, stream>>>(inf(idx), WH(dsto), n);
  };

  cvt(IN_ATTN_IN_W, o_h_ain, 3 * kD * kD);
  cvt(IN_ATTN_OUT_W, o_h_aout, kD * kD);
  cvt(IN_PM_DYN_W, o_h_dyn, 2 * kD * kDYN * kD);
  cvt(IN_PM_TDYN_W, o_h_tdyn, kP * kP * kD);
  cvt(IN_PM_OUT_W, o_h_pout, kD * 2 * kD * kP);
  cvt(IN_LIN1_W, o_h_lin1, kDFF * kD);
  cvt(IN_LIN2_W, o_h_lin2, kD * kDFF);
  cvt(IN_CLS_W, o_h_cls, kD * kD);
  cvt(IN_REG_W, o_h_reg, 3 * kD * kD);
  cvt(IN_LOGITS_W, o_h_logi, (kNC + 1) * kD);
  cvt(IN_DELTA_W, o_h_dlt, kP * kD);
  cvt(IN_PRO, o_proh, kB * kD);

  pd_proj_k<<<kB, 32, 0, stream>>>(inf(IN_PRO),
      inf(IN_PD_BOFF_W), inf(IN_PD_BOFF_B), inf(IN_PD_COFF_W), inf(IN_PD_COFF_B),
      inf(IN_PD_BW_W), inf(IN_PD_BW_B), inf(IN_PD_CW_W), inf(IN_PD_CW_B),
      WF(o_pdout));
  pd_sample_k<<<dim3(kB, kP), 256, 0, stream>>>(inf(IN_FEATURES), inf(IN_PROPOSAL),
      WF(o_pdout), WF(o_pf), WH(o_pfh));

  launch_gemm(WH(o_proh), WH(o_h_ain), inf(IN_ATTN_IN_B),
              nullptr, WH(o_qkvh), kB, 3 * kD, kD, 0, stream);
  attn_k<<<kN * kNH, 64, 0, stream>>>(WH(o_qkvh), WH(o_oah));
  launch_gemm(WH(o_oah), WH(o_h_aout), inf(IN_ATTN_OUT_B),
              WF(o_oproj), nullptr, kB, kD, kD, 0, stream);
  ln_row_k<<<kB, 256, 0, stream>>>(WF(o_oproj), inf(IN_PRO),
      inf(IN_NORM1_W), inf(IN_NORM1_B), WF(o_x1), WH(o_x1h), kD, kD, 0);

  launch_gemm(WH(o_x1h), WH(o_h_dyn), inf(IN_PM_DYN_B),
              nullptr, WH(o_dynh), kB, 2 * kD * kDYN, kD, 0, stream);
  launch_gemm(WH(o_x1h), WH(o_h_tdyn), inf(IN_PM_TDYN_B),
              WF(o_tdyn), nullptr, kB, kP * kP, kD, 0, stream);
  diff_gemm_k<<<kB, 256, 0, stream>>>(WF(o_tdyn), WF(o_pf), WF(o_G));
  diff_ln_k<<<kB, 256, 0, stream>>>(WF(o_G), inf(IN_PM_N4_W), inf(IN_PM_N4_B), WH(o_fcat));
  dyn_mix1_k<<<kB, 64, 0, stream>>>(WH(o_pfh), WH(o_dynh), WF(o_f1));
  ln_row_k<<<kB * kP, 256, 0, stream>>>(WF(o_f1), nullptr,
      inf(IN_PM_N1_W), inf(IN_PM_N1_B), nullptr, WH(o_f1ph), kDYN, kDYN, 1);
  dyn_mix2_k<<<kB, 256, 0, stream>>>(WH(o_f1ph), WH(o_dynh), WF(o_f2));
  ln_row_k<<<kB * kP, 256, 0, stream>>>(WF(o_f2), nullptr,
      inf(IN_PM_N2_W), inf(IN_PM_N2_B), nullptr, WH(o_fcat), kD, 2 * kD, 1);
  launch_gemm(WH(o_fcat), WH(o_h_pout), inf(IN_PM_OUT_B),
              WF(o_pmout), nullptr, kB, kD, 2 * kD * kP, 0, stream);
  ln_row_k<<<kB, 256, 0, stream>>>(WF(o_pmout), nullptr,
      inf(IN_PM_N3_W), inf(IN_PM_N3_B), WF(o_cfeat), nullptr, kD, kD, 1);

  ln_row_k<<<kB, 256, 0, stream>>>(WF(o_x1), WF(o_cfeat),
      inf(IN_NORM2_W), inf(IN_NORM2_B), WF(o_act1), WH(o_act1h), kD, kD, 0);
  launch_gemm(WH(o_act1h), WH(o_h_lin1), inf(IN_LIN1_B),
              nullptr, WH(o_hff), kB, kDFF, kD, 1, stream);
  launch_gemm(WH(o_hff), WH(o_h_lin2), inf(IN_LIN2_B),
              WF(o_ff), nullptr, kB, kD, kDFF, 0, stream);
  float* act_out = dout + (size_t)kB * (kNC + 1) + (size_t)kB * kP;
  ln_row_k<<<kB, 256, 0, stream>>>(WF(o_act1), WF(o_ff),
      inf(IN_NORM3_W), inf(IN_NORM3_B), act_out, WH(o_fch), kD, kD, 0);

  launch_gemm(WH(o_fch), WH(o_h_cls), nullptr, WF(o_clsr), nullptr, kB, kD, kD, 0, stream);
  ln_row_k<<<kB, 256, 0, stream>>>(WF(o_clsr), nullptr,
      inf(IN_CLS_LN_W), inf(IN_CLS_LN_B), nullptr, WH(o_clsfh), kD, kD, 1);
  size_t cur_h = o_fch;
  for (int i = 0; i < 3; ++i) {
    launch_gemm(WH(cur_h), WH(o_h_reg) + (size_t)i * kD * kD, nullptr,
                WF(o_regF), nullptr, kB, kD, kD, 0, stream);
    size_t nxt = (i & 1) ? o_regh1 : o_regh0;
    ln_row_k<<<kB, 256, 0, stream>>>(WF(o_regF), nullptr,
        inf(IN_REG_LN_W) + i * kD, inf(IN_REG_LN_B) + i * kD,
        nullptr, WH(nxt), kD, kD, 1);
    cur_h = nxt;
  }
  launch_gemm(WH(o_clsfh), WH(o_h_logi), inf(IN_LOGITS_B),
              dout, nullptr, kB, kNC + 1, kD, 0, stream);
  launch_gemm(WH(cur_h), WH(o_h_dlt), inf(IN_DELTA_B),
              WF(o_dlt), nullptr, kB, kP, kD, 0, stream);
  pred_k<<<(kB * kP + 255) / 256, 256, 0, stream>>>(inf(IN_PROPOSAL), WF(o_dlt),
      dout + (size_t)kB * (kNC + 1));
#undef WF
#undef WH
}
